// GroupedQueryAttention_45011257262697
// MI455X (gfx1250) — matrix operand order repaired and checked by reading, not compiled, not run on hardware
//
#include <hip/hip_runtime.h>
#include <math.h>

typedef unsigned short u16;
typedef unsigned int   u32;

typedef __bf16 v16bf __attribute__((ext_vector_type(16)));
typedef float  v8f   __attribute__((ext_vector_type(8)));
typedef int    vi4   __attribute__((vector_size(16)));   // matches builtin param type

// ---------------- problem constants ----------------
#define Mrows   16384          // B*S = 4*4096
#define HID     2048
#define NQKV    3072           // 2048 (Q) + 512 (K) + 512 (V)
#define BM      128
#define BN      128
#define BK      32
#define LDSS    40             // LDS row stride in halves (80B = 20-bank stride -> conflict-free b128)

// ---------------- bf16 hi/lo split helpers ----------------
__device__ __forceinline__ u16 f32_to_bf16_rne(float f) {
    u32 u = __float_as_uint(f);
    u32 r = u + 0x7FFFu + ((u >> 16) & 1u);
    return (u16)(r >> 16);
}
__device__ __forceinline__ void split_bf16(float f, u16& h, u16& l) {
    h = f32_to_bf16_rne(f);
    float fh = __uint_as_float(((u32)h) << 16);
    l = f32_to_bf16_rne(f - fh);
}

__device__ __forceinline__ v16bf load_frag(const u16* p0, const u16* p1) {
    union { uint4 q[2]; v16bf v; } u;
    u.q[0] = *(const uint4*)p0;
    u.q[1] = *(const uint4*)p1;
    return u.v;
}

// ---------------- gfx1250 async copy: global -> LDS, 16B per lane ----------------
__device__ __forceinline__ void async_cp16(const void* g, void* l) {
    __builtin_amdgcn_global_load_async_to_lds_b128(
        (__attribute__((address_space(1))) vi4*)g,
        (__attribute__((address_space(3))) vi4*)l,
        0, 0);
}
__device__ __forceinline__ void wait_async0() {
    asm volatile("s_wait_asynccnt 0x0" ::: "memory");
}

// ---------------- fp32 -> bf16 hi/lo split (weights + activations) ----------------
__global__ void split_f32_kernel(const float* __restrict__ src,
                                 u16* __restrict__ hi, u16* __restrict__ lo, int n) {
    int i = blockIdx.x * blockDim.x + threadIdx.x;
    if (i < n) { u16 h,l; split_bf16(src[i], h, l); hi[i] = h; lo[i] = l; }
}

__global__ void pack_bias_kernel(const float* __restrict__ bq, const float* __restrict__ bk,
                                 const float* __restrict__ bv, float* __restrict__ out) {
    int i = blockIdx.x * blockDim.x + threadIdx.x;   // 3072 total
    if (i < 2048)      out[i] = bq[i];
    else if (i < 2560) out[i] = bk[i - 2048];
    else if (i < 3072) out[i] = bv[i - 2560];
}

// ---------------- GEMM: C[M,N] = A[M,K] @ W[N,K]^T + bias ----------------
// A and W pre-split as bf16 hi/lo. 3-product bf16 emulation of fp32:
// hi*hi + hi*lo + lo*hi, fp32 WMMA accumulate. Tiles are copied global->LDS
// with GLOBAL_LOAD_ASYNC_TO_LDS_B128 (no staging VGPRs), double-buffered.
__global__ __launch_bounds__(256)
void gemm_bf16split_kernel(const u16* __restrict__ Ahi,
                           const u16* __restrict__ Alo,
                           const u16* __restrict__ Whi,
                           const u16* __restrict__ Wlo,
                           const float* __restrict__ bias,
                           float* __restrict__ C,
                           int M, int N, int K)
{
    __shared__ u16 lds[2][4][BM * LDSS];   // [buf][0:A_hi 1:A_lo 2:B_hi 3:B_lo]  (80 KB)

    const int tid  = threadIdx.x;
    const int lane = tid & 31;
    const int wave = tid >> 5;
    const int wm   = wave >> 2;       // 0..1 : 64-row slab
    const int wn   = wave & 3;        // 0..3 : 32-col slab
    const int lr   = lane & 15;
    const int lh   = lane >> 4;

    const int n0 = blockIdx.x * BN;
    const int m0 = blockIdx.y * BM;

    v8f acc[4][2];
    #pragma unroll
    for (int mt = 0; mt < 4; ++mt)
        #pragma unroll
        for (int nt = 0; nt < 2; ++nt)
            #pragma unroll
            for (int e = 0; e < 8; ++e) acc[mt][nt][e] = 0.0f;

    const int kTiles = K / BK;

    // async stage of one 128x32 bf16 tile per matrix: 512 b128 chunks, 2/thread
    auto stage = [&](int buf, int k0) {
        #pragma unroll
        for (int i = 0; i < 2; ++i) {
            int c   = tid + i * 256;
            int row = c >> 2, col = (c & 3) * 8;
            int lo  = row * LDSS + col;
            size_t ga = (size_t)(m0 + row) * K + k0 + col;
            size_t gb = (size_t)(n0 + row) * K + k0 + col;
            async_cp16(Ahi + ga, &lds[buf][0][lo]);
            async_cp16(Alo + ga, &lds[buf][1][lo]);
            async_cp16(Whi + gb, &lds[buf][2][lo]);
            async_cp16(Wlo + gb, &lds[buf][3][lo]);
        }
    };

    // ---- prologue ----
    stage(0, 0);
    wait_async0();
    __syncthreads();

    for (int kt = 0; kt < kTiles; ++kt) {
        const int cur = kt & 1, nxt = cur ^ 1;
        if (kt + 1 < kTiles) stage(nxt, (kt + 1) * BK);   // in flight during WMMAs

        // ---- compute current tile: 24 v_wmma_f32_16x16x32_bf16 per wave ----
        // B frag (32x16 KxN): lanes 0-15 hold K0..15 of col n=lr, lanes 16-31 K16..31
        v16bf bfh[2], bfl[2];
        #pragma unroll
        for (int nt = 0; nt < 2; ++nt) {
            int row  = wn * 32 + nt * 16 + lr;
            int base = row * LDSS + lh * 16;
            bfh[nt] = load_frag(&lds[cur][2][base], &lds[cur][2][base + 8]);
            bfl[nt] = load_frag(&lds[cur][3][base], &lds[cur][3][base + 8]);
        }
        #pragma unroll
        for (int mt = 0; mt < 4; ++mt) {
            // A frag (16x32 MxK): lane holds row lr, K lh*16..lh*16+15
            int row  = wm * 64 + mt * 16 + lr;
            int base = row * LDSS + lh * 16;
            v16bf afh = load_frag(&lds[cur][0][base], &lds[cur][0][base + 8]);
            v16bf afl = load_frag(&lds[cur][1][base], &lds[cur][1][base + 8]);
            #pragma unroll
            for (int nt = 0; nt < 2; ++nt) {
                acc[mt][nt] = __builtin_amdgcn_wmma_f32_16x16x32_bf16(
                    false, afh, false, bfh[nt], (short)0, acc[mt][nt], false, false);
                acc[mt][nt] = __builtin_amdgcn_wmma_f32_16x16x32_bf16(
                    false, afh, false, bfl[nt], (short)0, acc[mt][nt], false, false);
                acc[mt][nt] = __builtin_amdgcn_wmma_f32_16x16x32_bf16(
                    false, afl, false, bfh[nt], (short)0, acc[mt][nt], false, false);
            }
        }

        wait_async0();       // own async writes to buf[nxt] landed in LDS
        __syncthreads();     // everyone's landed; reads of buf[cur] done
    }

    // ---- epilogue: C/D layout — VGPR e holds M = e + 8*(lane>=16), N = lane&15 ----
    #pragma unroll
    for (int mt = 0; mt < 4; ++mt) {
        #pragma unroll
        for (int nt = 0; nt < 2; ++nt) {
            int n = n0 + wn * 32 + nt * 16 + lr;
            float bn = bias[n];
            #pragma unroll
            for (int e = 0; e < 8; ++e) {
                int m = m0 + wm * 64 + mt * 16 + e + lh * 8;
                C[(size_t)m * N + n] = acc[mt][nt][e] + bn;
            }
        }
    }
}

// ---------------- per-token GQA attention (memory-bound glue) ----------------
// One wave per (token m, group g). 4 queries x 4 keys x d=128.
// Writes the attention output directly as bf16 hi/lo for the O-projection GEMM.
__global__ __launch_bounds__(256)
void gqa_attn_kernel(const float* __restrict__ qkv,
                     u16* __restrict__ attn_hi, u16* __restrict__ attn_lo)
{
    const int lane = threadIdx.x & 31;
    const int wave = threadIdx.x >> 5;
    const int pair = blockIdx.x * 8 + wave;     // < 16384*4
    const int m = pair >> 2;
    const int g = pair & 3;

    const float* row = qkv + (size_t)m * NQKV;
    float q[4][4], k[4][4], v[4][4];
    #pragma unroll
    for (int i = 0; i < 4; ++i)
        #pragma unroll
        for (int c = 0; c < 4; ++c) {
            int d = c * 32 + lane;
            q[i][c] = row[g*512 + i*128 + d];
            k[i][c] = row[2048  + i*128 + d];
            v[i][c] = row[2560  + i*128 + d];
        }

    float s[4][4];
    #pragma unroll
    for (int i = 0; i < 4; ++i)
        #pragma unroll
        for (int j = 0; j < 4; ++j) {
            float p = q[i][0]*k[j][0] + q[i][1]*k[j][1]
                    + q[i][2]*k[j][2] + q[i][3]*k[j][3];
            #pragma unroll
            for (int off = 16; off > 0; off >>= 1) p += __shfl_xor(p, off, 32);
            s[i][j] = p * 0.08838834764831845f;   // 1/sqrt(128)
        }

    #pragma unroll
    for (int i = 0; i < 4; ++i) {
        float mx = fmaxf(fmaxf(s[i][0], s[i][1]), fmaxf(s[i][2], s[i][3]));
        float e0 = expf(s[i][0]-mx), e1 = expf(s[i][1]-mx);
        float e2 = expf(s[i][2]-mx), e3 = expf(s[i][3]-mx);
        float inv = 1.0f / (e0 + e1 + e2 + e3);
        #pragma unroll
        for (int c = 0; c < 4; ++c) {
            float o = (e0*v[0][c] + e1*v[1][c] + e2*v[2][c] + e3*v[3][c]) * inv;
            size_t idx = (size_t)m*HID + g*512 + i*128 + c*32 + lane;
            u16 h, l; split_bf16(o, h, l);
            attn_hi[idx] = h;
            attn_lo[idx] = l;
        }
    }
}

// ---------------- launch ----------------
extern "C" void kernel_launch(void* const* d_in, const int* in_sizes, int n_in,
                              void* d_out, int out_size, void* d_ws, size_t ws_size,
                              hipStream_t stream) {
    const float* x  = (const float*)d_in[0];
    const float* Wq = (const float*)d_in[1];
    const float* bq = (const float*)d_in[2];
    const float* Wk = (const float*)d_in[3];
    const float* bk = (const float*)d_in[4];
    const float* Wv = (const float*)d_in[5];
    const float* bv = (const float*)d_in[6];
    const float* Wo = (const float*)d_in[7];
    const float* bo = (const float*)d_in[8];
    float* out = (float*)d_out;

    char* ws = (char*)d_ws;
    size_t off = 0;
    auto alloc = [&](size_t bytes) -> void* {
        void* p = ws + off;
        off = (off + bytes + 255) & ~(size_t)255;
        return p;
    };
    u16*   x_hi    = (u16*)  alloc((size_t)Mrows * HID * 2);
    u16*   x_lo    = (u16*)  alloc((size_t)Mrows * HID * 2);
    u16*   wqkv_hi = (u16*)  alloc((size_t)NQKV * HID * 2);
    u16*   wqkv_lo = (u16*)  alloc((size_t)NQKV * HID * 2);
    u16*   wo_hi   = (u16*)  alloc((size_t)HID  * HID * 2);
    u16*   wo_lo   = (u16*)  alloc((size_t)HID  * HID * 2);
    float* bqkv    = (float*)alloc((size_t)NQKV * 4);
    float* qkvbuf  = (float*)alloc((size_t)Mrows * NQKV * 4);
    u16*   attn_hi = (u16*)  alloc((size_t)Mrows * HID * 2);
    u16*   attn_lo = (u16*)  alloc((size_t)Mrows * HID * 2);

    // 1) pre-split weights + activations into bf16 hi/lo
    split_f32_kernel<<<(2048*2048 + 255)/256, 256, 0, stream>>>(
        Wq, wqkv_hi,             wqkv_lo,             2048*2048);
    split_f32_kernel<<<(512*2048 + 255)/256, 256, 0, stream>>>(
        Wk, wqkv_hi + 2048*2048, wqkv_lo + 2048*2048, 512*2048);
    split_f32_kernel<<<(512*2048 + 255)/256, 256, 0, stream>>>(
        Wv, wqkv_hi + 2560*2048, wqkv_lo + 2560*2048, 512*2048);
    split_f32_kernel<<<(2048*2048 + 255)/256, 256, 0, stream>>>(
        Wo, wo_hi, wo_lo, 2048*2048);
    split_f32_kernel<<<(Mrows*HID + 255)/256, 256, 0, stream>>>(
        x, x_hi, x_lo, Mrows*HID);
    pack_bias_kernel<<<(NQKV + 255)/256, 256, 0, stream>>>(bq, bk, bv, bqkv);

    // 2) fused QKV projection: [16384,2048] x [2048,3072] -> qkv
    gemm_bf16split_kernel<<<dim3(NQKV/BN, Mrows/BM), 256, 0, stream>>>(
        x_hi, x_lo, wqkv_hi, wqkv_lo, bqkv, qkvbuf, Mrows, NQKV, HID);

    // 3) per-token grouped attention -> bf16 hi/lo
    gqa_attn_kernel<<<(Mrows * 4) / 8, 256, 0, stream>>>(qkvbuf, attn_hi, attn_lo);

    // 4) output projection: [16384,2048] x [2048,2048] -> d_out
    gemm_bf16split_kernel<<<dim3(HID/BN, Mrows/BM), 256, 0, stream>>>(
        attn_hi, attn_lo, wo_hi, wo_lo, bo, out, Mrows, HID, HID);
}
